// MultiHeadAttention_51316269252826
// MI455X (gfx1250) — matrix operand order repaired and checked by reading, not compiled, not run on hardware
//
#include <hip/hip_runtime.h>
#include <hip/hip_bf16.h>
#include <math.h>

// Problem constants (match reference)
#define B_   4
#define S_   2048
#define D_   1024
#define H_   16
#define DH_  64
#define LN_EPS 1e-5f

// ---------------------------------------------------------------------------
// WMMA types / helpers (gfx1250, wave32)
// ---------------------------------------------------------------------------
typedef __attribute__((ext_vector_type(16))) __bf16        v16bf;
typedef __attribute__((ext_vector_type(8)))  float         v8f;
typedef __attribute__((ext_vector_type(4)))  unsigned int  u32x4;
typedef __attribute__((ext_vector_type(4)))  float         f32x4;

union Frag {          // 16 bf16 per lane = one A or B fragment of 16x16x32
  v16bf v;
  u32x4 q[2];
};

__device__ __forceinline__ v8f wmma_bf16(const Frag& a, const Frag& b, v8f c) {
  // (neg_a, A, neg_b, B, c_mod, C, reuse_a, reuse_b)
  return __builtin_amdgcn_wmma_f32_16x16x32_bf16(false, a.v, false, b.v,
                                                 (short)0, c, false, false);
}

__device__ __forceinline__ v8f zero8() {
  v8f z;
#pragma unroll
  for (int i = 0; i < 8; ++i) z[i] = 0.0f;
  return z;
}

// ---- CDNA5 async global->LDS copy (ASYNCcnt path), one instr = 32 lanes x 16B
__device__ __forceinline__ void async_copy_b128(unsigned lds_byte_off,
                                                const void* gaddr) {
  asm volatile("global_load_async_to_lds_b128 %0, %1, off"
               :: "v"(lds_byte_off), "v"(gaddr) : "memory");
}
__device__ __forceinline__ void wait_async0() {
  asm volatile("s_wait_asynccnt 0" ::: "memory");
}
__device__ __forceinline__ unsigned lds_off(const void* p) {
  return (unsigned)(size_t)p;           // LDS aperture: addr[31:0] = LDS offset
}

// ---------------------------------------------------------------------------
// GEMM 1: C_bf16[M,N] = A_f32[M,K] @ W_f32[K,N] + bias
// block = 128 threads (4 waves), tile 128x64, K step 32
// each wave: 32 rows (2 A-frags) x 64 cols (4 B-frags) -> 8 WMMA / K-step
// ---------------------------------------------------------------------------
__global__ __launch_bounds__(128) void gemm_f32w_bf16out(
    const float* __restrict__ A, const float* __restrict__ W,
    const float* __restrict__ bias, __bf16* __restrict__ C,
    int M, int N, int K)
{
  __shared__ __align__(16) __bf16 at[128][40];  // [row][k], 80B row stride
  __shared__ __align__(16) __bf16 wt[64][40];   // [n][k] (transposed W tile)

  const int r0   = blockIdx.x * 128;
  const int n0   = blockIdx.y * 64;
  const int t    = threadIdx.x;
  const int wave = t >> 5;
  const int lane = t & 31;
  const int lrow = lane & 15;
  const int half = lane >> 4;

  v8f acc[2][4];
#pragma unroll
  for (int r = 0; r < 2; ++r)
#pragma unroll
    for (int s = 0; s < 4; ++s) acc[r][s] = zero8();

  for (int k0 = 0; k0 < K; k0 += 32) {
    __syncthreads();
    // one speculative prefetch of next W K-tile per thread (dropped if OOB)
    __builtin_prefetch(&W[(size_t)(k0 + 32 + (t >> 6)) * N + n0 + (t & 63)], 0, 1);
    // ---- stage A tile (128x32) f32 -> bf16, lane-coalesced float4 sweeps ----
#pragma unroll
    for (int st = 0; st < 8; ++st) {
      const int e4  = st * 128 + t;        // 1024 float4 chunks
      const int row = e4 >> 3;             // 8 float4 per 32-col row
      const int c4  = (e4 & 7) * 4;
      const float* src = A + (size_t)(r0 + row) * K + k0 + c4;
      __builtin_prefetch(src + 32, 0, 1);  // speculative: silently dropped OOB
      const f32x4 f = *(const f32x4*)src;
      __bf16* d = &at[row][c4];
      d[0] = (__bf16)f[0]; d[1] = (__bf16)f[1];
      d[2] = (__bf16)f[2]; d[3] = (__bf16)f[3];
    }
    // ---- stage W tile transposed wt[n][k], coalesced over n ----
#pragma unroll
    for (int st = 0; st < 16; ++st) {
      const int e = st * 128 + t;          // 2048 elems (32k x 64n)
      const int kq = e >> 6;
      const int n  = e & 63;
      wt[n][kq] = (__bf16)W[(size_t)(k0 + kq) * N + n0 + n];
    }
    __syncthreads();

    // ---- fragments: load everything first, then WMMA burst ----
    Frag a[2];
#pragma unroll
    for (int r = 0; r < 2; ++r) {
      const __bf16* ap = &at[wave * 32 + r * 16 + lrow][0];
      a[r].q[0] = *(const u32x4*)(ap + half * 16);
      a[r].q[1] = *(const u32x4*)(ap + half * 16 + 8);
    }
    Frag b[4];
#pragma unroll
    for (int s = 0; s < 4; ++s) {
      const __bf16* bp = &wt[s * 16 + lrow][half * 16];
      b[s].q[0] = *(const u32x4*)(bp);
      b[s].q[1] = *(const u32x4*)(bp + 8);
    }
#pragma unroll
    for (int s = 0; s < 4; ++s)
#pragma unroll
      for (int r = 0; r < 2; ++r)
        acc[r][s] = wmma_bf16(a[r], b[s], acc[r][s]);
  }

  // ---- epilogue: C-layout store, + bias ----
#pragma unroll
  for (int s = 0; s < 4; ++s) {
    const int col = n0 + s * 16 + lrow;
    const float bsv = bias[col];
#pragma unroll
    for (int r = 0; r < 2; ++r)
#pragma unroll
      for (int vv = 0; vv < 8; ++vv) {
        const int row = r0 + wave * 32 + r * 16 + vv + half * 8;
        C[(size_t)row * N + col] = (__bf16)(acc[r][s][vv] + bsv);
      }
  }
}

// ---------------------------------------------------------------------------
// GEMM 2: out_f32[M,N] = A_bf16[M,K] @ W_f32[K,N] + bias + resid
// A tile staged via CDNA5 async global->LDS DMA (no VGPR round-trip)
// ---------------------------------------------------------------------------
__global__ __launch_bounds__(128) void gemm_out_kernel(
    const __bf16* __restrict__ A, const float* __restrict__ W,
    const float* __restrict__ bias, const float* __restrict__ resid,
    float* __restrict__ C, int M, int N, int K)
{
  __shared__ __align__(16) __bf16 at[128][40];
  __shared__ __align__(16) __bf16 wt[64][40];

  const int r0   = blockIdx.x * 128;
  const int n0   = blockIdx.y * 64;
  const int t    = threadIdx.x;
  const int wave = t >> 5;
  const int lane = t & 31;
  const int lrow = lane & 15;
  const int half = lane >> 4;
  const unsigned atBase = lds_off(&at[0][0]);

  v8f acc[2][4];
#pragma unroll
  for (int r = 0; r < 2; ++r)
#pragma unroll
    for (int s = 0; s < 4; ++s) acc[r][s] = zero8();

  for (int k0 = 0; k0 < K; k0 += 32) {
    __syncthreads();
    __builtin_prefetch(&W[(size_t)(k0 + 32 + (t >> 6)) * N + n0 + (t & 63)], 0, 1);
    // ---- async-stage A tile (1024 x 16B chunks = 256 rows x 32 bf16) ----
#pragma unroll
    for (int i = 0; i < 8; ++i) {
      const int c   = i * 128 + t;         // chunk id, lane-coalesced
      const int row = c >> 2;              // 4 chunks per row
      const int ch  = c & 3;
      async_copy_b128(atBase + row * 80 + ch * 16,
                      A + (size_t)(r0 + row) * K + k0 + ch * 8);
    }
    // ---- stage W tile transposed (needs convert, stays on VALU path) ----
#pragma unroll
    for (int st = 0; st < 16; ++st) {
      const int e = st * 128 + t;
      const int kq = e >> 6;
      const int n  = e & 63;
      wt[n][kq] = (__bf16)W[(size_t)(k0 + kq) * N + n0 + n];
    }
    wait_async0();
    __syncthreads();

    Frag a[2];
#pragma unroll
    for (int r = 0; r < 2; ++r) {
      const __bf16* ap = &at[wave * 32 + r * 16 + lrow][0];
      a[r].q[0] = *(const u32x4*)(ap + half * 16);
      a[r].q[1] = *(const u32x4*)(ap + half * 16 + 8);
    }
    Frag b[4];
#pragma unroll
    for (int s = 0; s < 4; ++s) {
      const __bf16* bp = &wt[s * 16 + lrow][half * 16];
      b[s].q[0] = *(const u32x4*)(bp);
      b[s].q[1] = *(const u32x4*)(bp + 8);
    }
#pragma unroll
    for (int s = 0; s < 4; ++s)
#pragma unroll
      for (int r = 0; r < 2; ++r)
        acc[r][s] = wmma_bf16(a[r], b[s], acc[r][s]);
  }

#pragma unroll
  for (int s = 0; s < 4; ++s) {
    const int col = n0 + s * 16 + lrow;
    const float bsv = bias[col];
#pragma unroll
    for (int r = 0; r < 2; ++r)
#pragma unroll
      for (int vv = 0; vv < 8; ++vv) {
        const int row = r0 + wave * 32 + r * 16 + vv + half * 8;
        C[(size_t)row * N + col] = acc[r][s][vv] + bsv + resid[(size_t)row * N + col];
      }
  }
}

// ---------------------------------------------------------------------------
// Flash attention per head (head = contiguous [S, DH] slab due to the
// transpose-free reshape). Block = 128 threads (4 waves), 64-query tile,
// keys processed 32 at a time with online softmax. K tile is async-DMA'd
// into LDS (shared by all 4 waves); V tile is transposed manually.
// ---------------------------------------------------------------------------
__global__ __launch_bounds__(128) void attn_kernel(
    const __bf16* __restrict__ Qb, const __bf16* __restrict__ Kb,
    const __bf16* __restrict__ Vb, __bf16* __restrict__ Ob)
{
  const int bh = blockIdx.x;            // 0 .. B*H-1
  const int q0 = blockIdx.y * 64;       // query tile base
  const size_t hb = (size_t)bh * S_ * DH_;
  const __bf16* q = Qb + hb;
  const __bf16* k = Kb + hb;
  const __bf16* v = Vb + hb;
  __bf16*       o = Ob + hb;

  const int t    = threadIdx.x;
  const int wave = t >> 5;
  const int lane = t & 31;
  const int lrow = lane & 15;
  const int half = lane >> 4;

  __shared__ __align__(16) __bf16 kt[32][72];       // K tile, 144B row stride
  __shared__ __align__(16) __bf16 vt[64][40];       // V transposed: [dh][key]
  __shared__ __align__(16) __bf16 pb[4][16][40];    // per-wave P tile [q][key]
  const unsigned ktBase = lds_off(&kt[0][0]);

  // Q fragments (16 rows x 64 dh = 2 A-fragments), loaded once from global
  Frag qa[2];
  {
    const __bf16* qp = q + (size_t)(q0 + wave * 16 + lrow) * DH_;
#pragma unroll
    for (int d = 0; d < 2; ++d) {
      qa[d].q[0] = *(const u32x4*)(qp + d * 32 + half * 16);
      qa[d].q[1] = *(const u32x4*)(qp + d * 32 + half * 16 + 8);
    }
  }

  float m[8], l[8];
  v8f oacc[4];
#pragma unroll
  for (int vv = 0; vv < 8; ++vv) { m[vv] = -1e30f; l[vv] = 0.0f; }
#pragma unroll
  for (int s = 0; s < 4; ++s) oacc[s] = zero8();

  for (int kk = 0; kk < S_; kk += 32) {
    __syncthreads();                    // previous kt/vt fully consumed
    // ---- async-stage K tile: 32 keys x 64 dh = 256 x 16B chunks ----
#pragma unroll
    for (int i = 0; i < 2; ++i) {
      const int c   = i * 128 + t;
      const int row = c >> 3;           // 8 chunks per 64-elem key row
      const int ch  = c & 7;
      async_copy_b128(ktBase + row * 144 + ch * 16,
                      k + (size_t)(kk + row) * DH_ + ch * 8);
    }
    // ---- stage V tile transposed: vt[dh][key] ----
    {
      const int key = t >> 2;           // 0..31
      const int dh0 = (t & 3) * 16;
      const __bf16* src = v + (size_t)(kk + key) * DH_ + dh0;
      __builtin_prefetch(src + 32 * DH_, 0, 1);     // next V tile (speculative)
#pragma unroll
      for (int i = 0; i < 16; ++i) vt[dh0 + i][key] = src[i];
    }
    wait_async0();
    __syncthreads();

    // ---- scores: two 16x16 tiles (keys kk..kk+15, kk+16..kk+31) ----
    v8f st[2];
    st[0] = zero8(); st[1] = zero8();
#pragma unroll
    for (int j = 0; j < 2; ++j) {
#pragma unroll
      for (int d = 0; d < 2; ++d) {     // DH=64 -> 2 K-steps of 32
        Frag b;                         // B[dh][key]: key per lane, dh per half
        const __bf16* bp = &kt[j * 16 + lrow][d * 32 + half * 16];
        b.q[0] = *(const u32x4*)(bp);
        b.q[1] = *(const u32x4*)(bp + 8);
        st[j] = wmma_bf16(qa[d], b, st[j]);
      }
    }

    // ---- online softmax (row stats across 16-lane halves, wave32) ----
#pragma unroll
    for (int vv = 0; vv < 8; ++vv) {
      float s0 = st[0][vv] * 0.125f;    // 1/sqrt(64)
      float s1 = st[1][vv] * 0.125f;
      float tmax = fmaxf(s0, s1);
#pragma unroll
      for (int x = 1; x < 16; x <<= 1)
        tmax = fmaxf(tmax, __shfl_xor(tmax, x, 32));
      const float mn    = fmaxf(m[vv], tmax);
      const float p0    = __expf(s0 - mn);
      const float p1    = __expf(s1 - mn);
      const float alpha = __expf(m[vv] - mn);
      float rs = p0 + p1;
#pragma unroll
      for (int x = 1; x < 16; x <<= 1)
        rs += __shfl_xor(rs, x, 32);
      l[vv] = l[vv] * alpha + rs;
      m[vv] = mn;
#pragma unroll
      for (int s = 0; s < 4; ++s) oacc[s][vv] *= alpha;
      // C-layout -> LDS row-major P
      const int prow = vv + half * 8;
      pb[wave][prow][lrow]      = (__bf16)p0;
      pb[wave][prow][16 + lrow] = (__bf16)p1;
    }
    __syncthreads();                    // P visible to whole wave

    // ---- O += P(16x32) @ Vt(32x64) ----
    Frag pa;
    const __bf16* pp = &pb[wave][lrow][0];
    pa.q[0] = *(const u32x4*)(pp + half * 16);
    pa.q[1] = *(const u32x4*)(pp + half * 16 + 8);
    Frag b[4];
#pragma unroll
    for (int s = 0; s < 4; ++s) {
      const __bf16* bp = &vt[s * 16 + lrow][half * 16];
      b[s].q[0] = *(const u32x4*)(bp);
      b[s].q[1] = *(const u32x4*)(bp + 8);
    }
#pragma unroll
    for (int s = 0; s < 4; ++s)
      oacc[s] = wmma_bf16(pa, b[s], oacc[s]);
  }

  // ---- epilogue: O /= l, store bf16 ----
#pragma unroll
  for (int s = 0; s < 4; ++s) {
#pragma unroll
    for (int vv = 0; vv < 8; ++vv) {
      const int row = q0 + wave * 16 + vv + half * 8;
      const int col = s * 16 + lrow;
      o[(size_t)row * DH_ + col] = (__bf16)(oacc[s][vv] / l[vv]);
    }
  }
}

// ---------------------------------------------------------------------------
// LayerNorm in place over rows of D_=1024, one block (256 threads) per row
// ---------------------------------------------------------------------------
__global__ __launch_bounds__(256) void ln_kernel(
    float* __restrict__ X, const float* __restrict__ gamma,
    const float* __restrict__ beta)
{
  __shared__ float red[256];
  const int row = blockIdx.x;
  float* x = X + (size_t)row * D_;
  const int t = threadIdx.x;

  float v0[4];
  float s = 0.0f;
#pragma unroll
  for (int i = 0; i < 4; ++i) { v0[i] = x[t + i * 256]; s += v0[i]; }

  red[t] = s;
  __syncthreads();
  for (int off = 128; off > 0; off >>= 1) {
    if (t < off) red[t] += red[t + off];
    __syncthreads();
  }
  const float mu = red[0] * (1.0f / D_);
  __syncthreads();

  float vs = 0.0f;
#pragma unroll
  for (int i = 0; i < 4; ++i) { const float d = v0[i] - mu; vs += d * d; }
  red[t] = vs;
  __syncthreads();
  for (int off = 128; off > 0; off >>= 1) {
    if (t < off) red[t] += red[t + off];
    __syncthreads();
  }
  const float var = red[0] * (1.0f / D_);
  const float inv = rsqrtf(var + LN_EPS);
#pragma unroll
  for (int i = 0; i < 4; ++i) {
    const int c = t + i * 256;
    x[c] = (v0[i] - mu) * inv * gamma[c] + beta[c];
  }
}

// ---------------------------------------------------------------------------
// Launch
// ---------------------------------------------------------------------------
extern "C" void kernel_launch(void* const* d_in, const int* in_sizes, int n_in,
                              void* d_out, int out_size, void* d_ws, size_t ws_size,
                              hipStream_t stream)
{
  (void)in_sizes; (void)n_in; (void)out_size; (void)ws_size;
  const float* queries = (const float*)d_in[0];
  const float* keys    = (const float*)d_in[1];
  const float* values  = (const float*)d_in[2];
  const float* Wq      = (const float*)d_in[3];
  const float* bq      = (const float*)d_in[4];
  const float* Wk      = (const float*)d_in[5];
  const float* bk      = (const float*)d_in[6];
  const float* Wv      = (const float*)d_in[7];
  const float* bv      = (const float*)d_in[8];
  const float* Wo      = (const float*)d_in[9];
  const float* bo      = (const float*)d_in[10];
  const float* lng     = (const float*)d_in[11];
  const float* lnb     = (const float*)d_in[12];
  float* out = (float*)d_out;

  const int M = B_ * S_;                         // 8192
  __bf16* qbuf = (__bf16*)d_ws;                  // 16 MB each
  __bf16* kbuf = qbuf + (size_t)M * D_;
  __bf16* vbuf = kbuf + (size_t)M * D_;
  __bf16* abuf = vbuf + (size_t)M * D_;          // attention output (bf16)

  dim3 gg(M / 128, D_ / 64);
  gemm_f32w_bf16out<<<gg, 128, 0, stream>>>(queries, Wq, bq, qbuf, M, D_, D_);
  gemm_f32w_bf16out<<<gg, 128, 0, stream>>>(keys,    Wk, bk, kbuf, M, D_, D_);
  gemm_f32w_bf16out<<<gg, 128, 0, stream>>>(values,  Wv, bv, vbuf, M, D_, D_);

  attn_kernel<<<dim3(B_ * H_, S_ / 64), 128, 0, stream>>>(qbuf, kbuf, vbuf, abuf);

  gemm_out_kernel<<<gg, 128, 0, stream>>>(abuf, Wo, bo, queries, out, M, D_, D_);
  ln_kernel<<<M, 256, 0, stream>>>(out, lng, lnb);
}
